// Decoder_22402549416317
// MI455X (gfx1250) — matrix operand order repaired and checked by reading, not compiled, not run on hardware
//
#include <hip/hip_runtime.h>

#define B_    32
#define L_    48
#define LP_   49
#define D_    256
#define H_    8
#define DH_   32
#define DFF_  1024
#define SMAX_ 64   // padded KV-cache sequence capacity

typedef __attribute__((ext_vector_type(16))) _Float16 v16h;
typedef __attribute__((ext_vector_type(8)))  _Float16 v8h;
typedef __attribute__((ext_vector_type(8)))  float    v8f;

// ---------------- weight prep: W[K,N] f32 -> WT[N,K] f16 (row = output column) ---
__global__ void k_transpose_f16(const float* __restrict__ W, _Float16* __restrict__ WT,
                                int K, int N) {
  int idx = blockIdx.x * 256 + threadIdx.x;
  if (idx >= K * N) return;
  int n = idx / K, k = idx - n * K;
  WT[idx] = (_Float16)W[(size_t)k * N + n];
}

// plain f32 -> f16 conversion (for enc_output)
__global__ void k_cvt_f16(const float* __restrict__ in, _Float16* __restrict__ o, int n) {
  int idx = blockIdx.x * 256 + threadIdx.x;
  if (idx < n) o[idx] = (_Float16)in[idx];
}

// concat three 256-element biases -> one 768-element bias (q|k|v)
__global__ void k_concat3(const float* __restrict__ a, const float* __restrict__ b,
                          const float* __restrict__ c, float* __restrict__ o) {
  int i = threadIdx.x, g = blockIdx.x;
  o[g * 256 + i] = (g == 0) ? a[i] : (g == 1) ? b[i] : c[i];
}

__global__ void k_init_tokens(int* tokens) {
  int b = threadIdx.x;
  if (b < B_) tokens[b * LP_] = 0;  // SOS_CODE
}

// x[b,:] = emb[token]*sqrt(D) + positional_encoding(t,:)   (f32 + f16 copies)
__global__ __launch_bounds__(256)
void k_embed(const float* __restrict__ emb, const int* __restrict__ tokens,
             float* __restrict__ x, _Float16* __restrict__ xh, int t) {
  int b = blockIdx.x, i = threadIdx.x;
  int tok = tokens[b * LP_ + t];
  float fi  = (float)((i >> 1) << 1);
  float ang = (float)t * expf(-logf(10000.0f) * fi / (float)D_);
  float pe  = (i & 1) ? cosf(ang) : sinf(ang);
  float v   = emb[(size_t)tok * D_ + i] * 16.0f + pe;   // sqrt(256)=16
  x[b * D_ + i]  = v;
  xh[b * D_ + i] = (_Float16)v;
}

// ---------------- WMMA GEMM: C[M,N] = A[M,K](f16) @ WT^T (WT=[N,K] f16) + bias ----
// One wave per 16x16 tile; pure 16B loads for both fragments.
//   mode 0: C f32 [m*ldc+n]
//   mode 1: KV-cache store [B,H,SMAX,DH]: row m -> (b=m/R, s=pos+m%R)
//   mode 2: fused QKV (N=768): cols 0-255 -> C (qbuf), 256-511 -> cacheK@pos,
//           512-767 -> cacheV@pos (R==1)
//   mode 3: Ch f16 [m*ldc+n] (+ optional relu)  -- FFN1 -> FFN2 chaining
__global__ __launch_bounds__(32)
void k_gemm(const _Float16* __restrict__ A, int lda,
            const _Float16* __restrict__ WT,
            const float* __restrict__ bias,
            float* __restrict__ C, _Float16* __restrict__ Ch, int ldc,
            int K, int relu, int mode,
            float* __restrict__ cacheK, float* __restrict__ cacheV,
            int R, int posBase)
{
  const int nt = blockIdx.x, mt = blockIdx.y;
  const int lane = threadIdx.x;
  const int lx = lane & 15;
  const int hf = lane >> 4;          // half-wave select
  const int m  = mt * 16 + lx;       // A row for this lane
  const int n  = nt * 16 + lx;       // B/C column for this lane

  const _Float16* arow = A  + (size_t)m * lda;
  const _Float16* brow = WT + (size_t)n * K;

  v8f acc = {};
  for (int k0 = 0; k0 < K; k0 += 32) {
    const v8h* ap = (const v8h*)(arow + k0 + hf * 8);
    v8h a0 = ap[0];                   // K = hf*8 + 0..7
    v8h a1 = ap[2];                   // K = hf*8 + 16..23
    const v8h* bp = (const v8h*)(brow + k0 + hf * 8);
    v8h b0 = bp[0], b1 = bp[2];       // K = hf*8 + 0..7, hf*8 + 16..23
    v16h a, b;
#pragma unroll
    for (int i = 0; i < 8; i++) { a[i] = a0[i]; a[8 + i] = a1[i];
                                  b[i] = b0[i]; b[8 + i] = b1[i]; }
    acc = __builtin_amdgcn_wmma_f32_16x16x32_f16(false, a, false, b,
                                                 (short)0, acc, false, false);
  }

  float bv = bias ? bias[n] : 0.0f;
#pragma unroll
  for (int v = 0; v < 8; v++) {                 // C layout: row = mt*16 + hf*8 + v, col = n
    int   mm  = mt * 16 + hf * 8 + v;
    float val = acc[v] + bv;
    if (relu) val = fmaxf(val, 0.0f);
    if (mode == 0) {
      C[(size_t)mm * ldc + n] = val;
    } else if (mode == 3) {
      Ch[(size_t)mm * ldc + n] = (_Float16)val;
    } else if (mode == 1) {
      int b_ = mm / R, s = posBase + (mm - b_ * R);
      cacheK[(((size_t)b_ * H_ + (n >> 5)) * SMAX_ + s) * DH_ + (n & 31)] = val;
    } else {                                    // mode 2 (R==1, batch == row)
      int g = n >> 8, nn = n & 255;
      if (g == 0) {
        C[(size_t)mm * ldc + nn] = val;
      } else {
        float* cc = (g == 1) ? cacheK : cacheV;
        cc[(((size_t)mm * H_ + (nn >> 5)) * SMAX_ + posBase) * DH_ + (nn & 31)] = val;
      }
    }
  }
}

// ---------------- single-query attention vs KV cache (one block per (b,h)) --------
__global__ __launch_bounds__(64)
void k_attn(const float* __restrict__ q, const float* __restrict__ Kc,
            const float* __restrict__ Vc, _Float16* __restrict__ out, int seqlen)
{
  __shared__ float sl[SMAX_];
  int b = blockIdx.x / H_, h = blockIdx.x % H_;
  int tid = threadIdx.x;
  const float scale = 0.17677669529663687f;   // 1/sqrt(32)
  const float* qv = q + b * D_ + h * DH_;
  const float* Kb = Kc + ((size_t)(b * H_ + h)) * SMAX_ * DH_;
  if (tid < seqlen) {
    const float* kr = Kb + tid * DH_;
    float acc = 0.f;
#pragma unroll
    for (int d = 0; d < DH_; d++) acc += qv[d] * kr[d];
    sl[tid] = acc * scale;
  } else sl[tid] = -1e30f;
  __syncthreads();
  float mx = -1e30f;
  for (int s = 0; s < seqlen; s++) mx = fmaxf(mx, sl[s]);
  float sum = 0.f;
  for (int s = 0; s < seqlen; s++) sum += expf(sl[s] - mx);
  float inv = 1.0f / sum;
  if (tid < DH_) {
    const float* Vb = Vc + ((size_t)(b * H_ + h)) * SMAX_ * DH_ + tid;
    float acc = 0.f;
    for (int s = 0; s < seqlen; s++) acc += expf(sl[s] - mx) * Vb[s * DH_];
    out[b * D_ + h * DH_ + tid] = (_Float16)(acc * inv);
  }
}

// ---------------- pointer head: mean-over-heads logits -> softmax row + argmax ----
__global__ __launch_bounds__(64)
void k_pointer(const float* __restrict__ q, const float* __restrict__ lastK,
               const int* __restrict__ enc_input, int* __restrict__ tokens,
               float* __restrict__ out, int t)
{
  __shared__ float sl[SMAX_];
  int b = blockIdx.x, s = threadIdx.x;
  const float cscale = 0.17677669529663687f / 8.0f;  // (1/sqrt(dh)) / num_heads
  if (s < L_) {
    float acc = 0.f;
    for (int h = 0; h < H_; h++) {
      const float* qv = q + b * D_ + h * DH_;
      const float* kr = lastK + (((size_t)(b * H_ + h)) * SMAX_ + s) * DH_;
#pragma unroll
      for (int d = 0; d < DH_; d++) acc += qv[d] * kr[d];
    }
    sl[s] = acc * cscale;
  } else sl[s] = -1e30f;
  __syncthreads();
  float mx = -1e30f; for (int k = 0; k < L_; k++) mx = fmaxf(mx, sl[k]);
  float sum = 0.f;   for (int k = 0; k < L_; k++) sum += expf(sl[k] - mx);
  if (s < L_) out[((size_t)b * L_ + t) * L_ + s] = expf(sl[s] - mx) / sum;
  if (s == 0) {
    float best = sl[0]; int bi = 0;
    for (int k = 1; k < L_; k++) if (sl[k] > best) { best = sl[k]; bi = k; }
    if (t + 1 < LP_) tokens[b * LP_ + t + 1] = enc_input[b * L_ + bi];
  }
}

// ---------------- y = LayerNorm(x + r)  (f32 + f16 copies) ------------------------
__global__ __launch_bounds__(256)
void k_ln(const float* __restrict__ x, const float* __restrict__ r,
          const float* __restrict__ g, const float* __restrict__ be,
          float* __restrict__ y, _Float16* __restrict__ yh)
{
  __shared__ float sm[256];
  int row = blockIdx.x, i = threadIdx.x;
  float h = x[row * D_ + i] + r[row * D_ + i];
  sm[i] = h; __syncthreads();
  for (int off = 128; off > 0; off >>= 1) { if (i < off) sm[i] += sm[i + off]; __syncthreads(); }
  float mean = sm[0] * (1.0f / D_); __syncthreads();
  float dv = h - mean;
  sm[i] = dv * dv; __syncthreads();
  for (int off = 128; off > 0; off >>= 1) { if (i < off) sm[i] += sm[i + off]; __syncthreads(); }
  float var = sm[0] * (1.0f / D_);
  float o = g[i] * dv * rsqrtf(var + 1e-6f) + be[i];
  y[row * D_ + i]  = o;
  yh[row * D_ + i] = (_Float16)o;
}

// ==================================================================================
extern "C" void kernel_launch(void* const* d_in, const int* in_sizes, int n_in,
                              void* d_out, int out_size, void* d_ws, size_t ws_size,
                              hipStream_t stream)
{
  (void)in_sizes; (void)out_size; (void)ws_size;
  if (n_in < 72 || !d_out || !d_ws) return;

  auto F = [&](int i) { return (const float*)d_in[i]; };
  const int*   enc_input  = (const int*)d_in[0];
  const float* enc_output = F(1);
  const float* embedding  = F(4);
  // params["last"]  (sorted: ln1{b,g}, mha1{wk{b,w},wo{b,w},wq{b,w},wv{b,w}}, wk{b,w}, wq{b,w})
  const float *Lln_b = F(5), *Lln_g = F(6);
  const float *Lm_kb = F(7),  *Lm_kw = F(8),  *Lm_ob = F(9),  *Lm_ow = F(10),
              *Lm_qb = F(11), *Lm_qw = F(12), *Lm_vb = F(13), *Lm_vw = F(14);
  const float *Lwk_b = F(15), *Lwk_w = F(16), *Lwq_b = F(17), *Lwq_w = F(18);
  // params["layers"][l]  (sorted: ffn1,ffn2,ln1,ln2,ln3,mha1,mha2)
  struct Lyr {
    const float *f1b,*f1w,*f2b,*f2w,*l1b,*l1g,*l2b,*l2g,*l3b,*l3g;
    const float *m1kb,*m1kw,*m1ob,*m1ow,*m1qb,*m1qw,*m1vb,*m1vw;
    const float *m2kb,*m2kw,*m2ob,*m2ow,*m2qb,*m2qw,*m2vb,*m2vw;
  } Lr[2];
  for (int l = 0; l < 2; l++) {
    int b = 19 + l * 26;
    Lr[l] = { F(b+0),F(b+1),F(b+2),F(b+3),F(b+4),F(b+5),F(b+6),F(b+7),F(b+8),F(b+9),
              F(b+10),F(b+11),F(b+12),F(b+13),F(b+14),F(b+15),F(b+16),F(b+17),
              F(b+18),F(b+19),F(b+20),F(b+21),F(b+22),F(b+23),F(b+24),F(b+25) };
  }

  // ---- workspace sub-allocator ----
  char* wsb = (char*)d_ws; size_t off = 0;
  auto alloc = [&](size_t bytes) -> void* {
    void* p = wsb + off; off = (off + bytes + 255) & ~(size_t)255; return p;
  };
  auto prep = [&](const float* W, int K, int N) -> _Float16* {
    _Float16* p = (_Float16*)alloc((size_t)K * N * sizeof(_Float16));
    int tot = K * N;
    k_transpose_f16<<<(tot + 255) / 256, 256, 0, stream>>>(W, p, K, N);
    return p;
  };
  // fused QKV: WT rows 0-255 = wq cols, 256-511 = wk, 512-767 = wv; concat bias
  auto prep_qkv = [&](const float* wq, const float* wk, const float* wv,
                      const float* bq, const float* bk, const float* bv,
                      _Float16** wt, float** bb) {
    _Float16* p = (_Float16*)alloc((size_t)3 * D_ * D_ * sizeof(_Float16));
    int tot = D_ * D_;
    k_transpose_f16<<<(tot + 255) / 256, 256, 0, stream>>>(wq, p,             D_, D_);
    k_transpose_f16<<<(tot + 255) / 256, 256, 0, stream>>>(wk, p + tot,       D_, D_);
    k_transpose_f16<<<(tot + 255) / 256, 256, 0, stream>>>(wv, p + 2 * tot,   D_, D_);
    float* cb = (float*)alloc(3 * D_ * sizeof(float));
    k_concat3<<<3, 256, 0, stream>>>(bq, bk, bv, cb);
    *wt = p; *bb = cb;
  };
  auto cachef = [&]() -> float* {
    return (float*)alloc((size_t)B_ * H_ * SMAX_ * DH_ * sizeof(float));
  };
  auto gemm = [&](const _Float16* A, int lda, const _Float16* WT, const float* bias,
                  float* C, _Float16* Ch, int ldc, int M, int N, int K, int relu,
                  int mode, float* cK, float* cV, int R, int pos) {
    dim3 g(N / 16, M / 16);
    k_gemm<<<g, 32, 0, stream>>>(A, lda, WT, bias, C, Ch, ldc, K, relu, mode, cK, cV, R, pos);
  };

  // ---- fp16 weights (index 2 of self arrays = last.mha1) ----
  _Float16 *Wqkv[3], *Wo1[3], *Wq2[2], *Wk2[2], *Wv2[2], *Wo2[2], *Wf1[2], *Wf2[2];
  float    *Bqkv[3];
  const float *ob1[3];
  for (int l = 0; l < 2; l++) {
    prep_qkv(Lr[l].m1qw, Lr[l].m1kw, Lr[l].m1vw,
             Lr[l].m1qb, Lr[l].m1kb, Lr[l].m1vb, &Wqkv[l], &Bqkv[l]);
    Wo1[l] = prep(Lr[l].m1ow, D_, D_);
    Wq2[l] = prep(Lr[l].m2qw, D_, D_);  Wk2[l] = prep(Lr[l].m2kw, D_, D_);
    Wv2[l] = prep(Lr[l].m2vw, D_, D_);  Wo2[l] = prep(Lr[l].m2ow, D_, D_);
    Wf1[l] = prep(Lr[l].f1w, D_, DFF_); Wf2[l] = prep(Lr[l].f2w, DFF_, D_);
    ob1[l] = Lr[l].m1ob;
  }
  prep_qkv(Lm_qw, Lm_kw, Lm_vw, Lm_qb, Lm_kb, Lm_vb, &Wqkv[2], &Bqkv[2]);
  Wo1[2] = prep(Lm_ow, D_, D_);  ob1[2] = Lm_ob;
  _Float16* WLq = prep(Lwq_w, D_, D_);
  _Float16* WLk = prep(Lwk_w, D_, D_);

  // ---- KV caches + activations ----
  float *selfK[3], *selfV[3], *crossK[2], *crossV[2];
  for (int i = 0; i < 3; i++) { selfK[i] = cachef(); selfV[i] = cachef(); }
  for (int i = 0; i < 2; i++) { crossK[i] = cachef(); crossV[i] = cachef(); }
  float* lastK = cachef();
  float*     xbuf  = (float*)alloc(B_ * D_ * sizeof(float));
  _Float16*  xh    = (_Float16*)alloc(B_ * D_ * sizeof(_Float16));
  float*     qbuf  = (float*)alloc(B_ * D_ * sizeof(float));
  _Float16*  attnh = (_Float16*)alloc(B_ * D_ * sizeof(_Float16));
  float*     tmpb  = (float*)alloc(B_ * D_ * sizeof(float));
  _Float16*  ffnmh = (_Float16*)alloc(B_ * DFF_ * sizeof(_Float16));
  _Float16*  ench  = (_Float16*)alloc((size_t)B_ * L_ * D_ * sizeof(_Float16));
  int*       tokens = (int*)alloc(B_ * LP_ * sizeof(int));

  // ---- one-time prep: enc_output f16, cross K/V, pointer K, token init ----
  {
    int tot = B_ * L_ * D_;
    k_cvt_f16<<<(tot + 255) / 256, 256, 0, stream>>>(enc_output, ench, tot);
  }
  for (int l = 0; l < 2; l++) {
    gemm(ench, D_, Wk2[l], Lr[l].m2kb, nullptr, nullptr, 0, B_*L_, D_, D_, 0, 1, crossK[l], nullptr, L_, 0);
    gemm(ench, D_, Wv2[l], Lr[l].m2vb, nullptr, nullptr, 0, B_*L_, D_, D_, 0, 1, crossV[l], nullptr, L_, 0);
  }
  gemm(ench, D_, WLk, Lwk_b, nullptr, nullptr, 0, B_*L_, D_, D_, 0, 1, lastK, nullptr, L_, 0);
  k_init_tokens<<<1, 32, 0, stream>>>(tokens);

  // ---- incremental rollout: each step emits one output row of comb ----
  for (int t = 0; t < L_; t++) {
    k_embed<<<B_, 256, 0, stream>>>(embedding, tokens, xbuf, xh, t);
    for (int l = 0; l < 2; l++) {
      // self-attention: fused QKV (q -> qbuf, K/V appended at position t)
      gemm(xh, D_, Wqkv[l], Bqkv[l], qbuf, nullptr, D_, B_, 3*D_, D_, 0, 2, selfK[l], selfV[l], 1, t);
      k_attn<<<B_*H_, 64, 0, stream>>>(qbuf, selfK[l], selfV[l], attnh, t + 1);
      gemm(attnh, D_, Wo1[l], ob1[l], tmpb, nullptr, D_, B_, D_, D_, 0, 0, nullptr, nullptr, 1, 0);
      k_ln<<<B_, 256, 0, stream>>>(xbuf, tmpb, Lr[l].l1g, Lr[l].l1b, xbuf, xh);
      // cross-attention (precomputed K/V)
      gemm(xh, D_, Wq2[l], Lr[l].m2qb, qbuf, nullptr, D_, B_, D_, D_, 0, 0, nullptr, nullptr, 1, 0);
      k_attn<<<B_*H_, 64, 0, stream>>>(qbuf, crossK[l], crossV[l], attnh, L_);
      gemm(attnh, D_, Wo2[l], Lr[l].m2ob, tmpb, nullptr, D_, B_, D_, D_, 0, 0, nullptr, nullptr, 1, 0);
      k_ln<<<B_, 256, 0, stream>>>(xbuf, tmpb, Lr[l].l2g, Lr[l].l2b, xbuf, xh);
      // FFN (f16 intermediate, relu fused)
      gemm(xh, D_, Wf1[l], Lr[l].f1b, nullptr, ffnmh, DFF_, B_, DFF_, D_, 1, 3, nullptr, nullptr, 1, 0);
      gemm(ffnmh, DFF_, Wf2[l], Lr[l].f2b, tmpb, nullptr, D_, B_, D_, DFF_, 0, 0, nullptr, nullptr, 1, 0);
      k_ln<<<B_, 256, 0, stream>>>(xbuf, tmpb, Lr[l].l3g, Lr[l].l3b, xbuf, xh);
    }
    // "last" block: self-MHA + LN, then pointer attention
    gemm(xh, D_, Wqkv[2], Bqkv[2], qbuf, nullptr, D_, B_, 3*D_, D_, 0, 2, selfK[2], selfV[2], 1, t);
    k_attn<<<B_*H_, 64, 0, stream>>>(qbuf, selfK[2], selfV[2], attnh, t + 1);
    gemm(attnh, D_, Wo1[2], ob1[2], tmpb, nullptr, D_, B_, D_, D_, 0, 0, nullptr, nullptr, 1, 0);
    k_ln<<<B_, 256, 0, stream>>>(xbuf, tmpb, Lln_g, Lln_b, xbuf, xh);
    gemm(xh, D_, WLq, Lwq_b, qbuf, nullptr, D_, B_, D_, D_, 0, 0, nullptr, nullptr, 1, 0);
    k_pointer<<<B_, 64, 0, stream>>>(qbuf, lastK, enc_input, tokens, (float*)d_out, t);
  }
}
